// GraphCLIP_57518202028157
// MI455X (gfx1250) — matrix operand order repaired and checked by reading, not compiled, not run on hardware
//
#include <hip/hip_runtime.h>
#include <math.h>
#include <stdint.h>

// ---------------- problem constants ----------------
#define L_LAYERS 5
#define H 256
#define H4 1024
#define NN 200000
#define EE 400000
#define GG 8000
#define NPG 25   // nodes per graph (batch = i / 25, sorted)

typedef __bf16 bf16;
typedef bf16  v16bf __attribute__((ext_vector_type(16)));
typedef bf16  bf16x8 __attribute__((ext_vector_type(8)));
typedef float v8f   __attribute__((ext_vector_type(8)));

__device__ __forceinline__ float gelu_exact(float x) {
    return 0.5f * x * (1.0f + erff(x * 0.70710678118654752440f));
}

// ---------------------------------------------------------------------------
// Pack a row-major fp32 weight [K, Nout] into the per-lane WMMA B-fragment
// layout (16-bit B 32x16 per tile; contiguous k order used by this file, matching the A fragment):
//   lanes 0-15  : col = cg*16 + lane,     K = kt*32 + e      (e = 0..15)
//   lanes 16-31 : col = cg*16 + lane-16,  K = kt*32 + 16 + e
// Blob index = ((kt * (Nout/16) + cg) * 32 + lane) * 16 bf16 -> one contiguous
// 32-byte load per lane per fragment.
// ---------------------------------------------------------------------------
__global__ void pack_b_kernel(const float* __restrict__ W, bf16* __restrict__ out,
                              int K, int Nout) {
    int idx = blockIdx.x * blockDim.x + threadIdx.x;
    int total = (K / 32) * (Nout / 16) * 32;
    if (idx >= total) return;
    int lane = idx & 31;
    int cg   = (idx >> 5) % (Nout / 16);
    int kt   = (idx >> 5) / (Nout / 16);
    int col  = cg * 16 + (lane & 15);
    int kb   = kt * 32 + ((lane >> 4) * 16);
    bf16* o = out + (size_t)idx * 16;
#pragma unroll
    for (int e = 0; e < 16; ++e)
        o[e] = (bf16)W[(size_t)(kb + e) * Nout + col];
}

// h = atom_emb[x]
__global__ void init_h_kernel(const int* __restrict__ x, const float* __restrict__ atom_emb,
                              float* __restrict__ h) {
    int i = blockIdx.x, c = threadIdx.x;
    h[(size_t)i * H + c] = atom_emb[(size_t)x[i] * H + c];
}

__global__ void init_vn_kernel(const float* __restrict__ vn_emb, float* __restrict__ vn) {
    vn[(size_t)blockIdx.x * H + threadIdx.x] = vn_emb[threadIdx.x];
}

// h_in = h + vn[batch];  aggr = 0
__global__ void addvn_kernel(const float* __restrict__ h, const float* __restrict__ vn,
                             const int* __restrict__ batch,
                             float* __restrict__ h_in, float* __restrict__ aggr) {
    int i = blockIdx.x, c = threadIdx.x;
    size_t o = (size_t)i * H + c;
    h_in[o] = h[o] + vn[(size_t)batch[i] * H + c];
    aggr[o] = 0.0f;
}

// per edge: aggr[dst] += gelu(h_in[src] + bond_emb[edge_attr])
__global__ void edge_kernel(const float* __restrict__ h_in, const int* __restrict__ ei,
                            const int* __restrict__ ea, const float* __restrict__ bond,
                            float* __restrict__ aggr) {
    int e = blockIdx.x, c = threadIdx.x;
    int s = ei[e], d = ei[EE + e], t = ea[e];
    float v = h_in[(size_t)s * H + c] + bond[(size_t)t * H + c];
    v = gelu_exact(v);
    atomicAdd(&aggr[(size_t)d * H + c], v);
}

// xb = bf16((1+eps[l])*h_in + aggr)
__global__ void prepx_kernel(const float* __restrict__ h_in, const float* __restrict__ aggr,
                             const float* __restrict__ eps, int l, bf16* __restrict__ xb) {
    int i = blockIdx.x, c = threadIdx.x;
    size_t o = (size_t)i * H + c;
    xb[o] = (bf16)((1.0f + eps[l]) * h_in[o] + aggr[o]);
}

// vpb[g] = bf16(max over 25 nodes of h_in)
__global__ void segmax_kernel(const float* __restrict__ h_in, bf16* __restrict__ vpb) {
    int g = blockIdx.x, c = threadIdx.x;
    const float* p = h_in + (size_t)g * NPG * H + c;
    float m = -3.4e38f;
#pragma unroll
    for (int j = 0; j < NPG; ++j) m = fmaxf(m, p[(size_t)j * H]);
    vpb[(size_t)g * H + c] = (bf16)m;
}

// hgb[g] = bf16(sum over 25 nodes of h)
__global__ void segsum_kernel(const float* __restrict__ h, bf16* __restrict__ hgb) {
    int g = blockIdx.x, c = threadIdx.x;
    const float* p = h + (size_t)g * NPG * H + c;
    float s = 0.0f;
#pragma unroll
    for (int j = 0; j < NPG; ++j) s += p[(size_t)j * H];
    hgb[(size_t)g * H + c] = (bf16)s;
}

// out = z / ||z||
__global__ void l2norm_kernel(const float* __restrict__ z, float* __restrict__ out) {
    int g = blockIdx.x, c = threadIdx.x;
    float v = z[(size_t)g * H + c];
    __shared__ float ssq[256];
    ssq[c] = v * v;
    __syncthreads();
    for (int s = 128; s > 0; s >>= 1) {
        if (c < s) ssq[c] += ssq[c + s];
        __syncthreads();
    }
    out[(size_t)g * H + c] = v * rsqrtf(ssq[0]);
}

// ---------------------------------------------------------------------------
// Fused WMMA GEMM:  D = epilogue(A[M,K](bf16) @ Bpacked[K,NOUT] + bias)
// Block = 16 rows x NOUT cols (full rows => in-kernel LayerNorm).
// 8 wave32s, each owning NACC 16x16 v_wmma_f32_16x16x32_bf16 tiles.
// A tile staged into LDS with CDNA5 async global->LDS copies (ASYNCcnt).
// Epilogue order: +bias -> (LN g,b) -> (GELU) -> (+resid) -> store f32/bf16.
// ---------------------------------------------------------------------------
template <int NACC>
__launch_bounds__(256)
__global__ void fused_gemm(const bf16* __restrict__ A, const bf16* __restrict__ Bp,
                           const float* __restrict__ bias,
                           const float* __restrict__ lng, const float* __restrict__ lnb,
                           int do_gelu,
                           const float* __restrict__ resid,
                           float* __restrict__ out_f32, bf16* __restrict__ out_bf16,
                           int K) {
    constexpr int NOUT = NACC * 16 * 8;
    __shared__ bf16 Atile[16 * 1024];
    __shared__ float red[2][16][8];

    const int row0 = blockIdx.x * 16;
    const int tid  = threadIdx.x;
    const int wave = tid >> 5;
    const int lane = tid & 31;

    // ---- A tile stage: direct global->LDS async copies (16B per lane) ----
    {
        const int elems = 16 * K;  // bf16 elements; 16B-aligned chunks of 8
        for (int i = tid * 8; i < elems; i += 256 * 8) {
            unsigned lds_addr = (unsigned)(uintptr_t)(&Atile[i]);
            const bf16* gp = &A[(size_t)row0 * K + i];
            asm volatile("global_load_async_to_lds_b128 %0, %1, off"
                         :: "v"(lds_addr), "v"(gp)
                         : "memory");
        }
        asm volatile("s_wait_asynccnt 0" ::: "memory");
    }
    __syncthreads();

    v8f acc[NACC];
    const v8f vzero = {0, 0, 0, 0, 0, 0, 0, 0};
#pragma unroll
    for (int a = 0; a < NACC; ++a) acc[a] = vzero;

    const int colbase = wave * (NACC * 16);
    const int arow = lane & 15;
    const int koff = (lane >> 4) * 16;

    for (int k0 = 0; k0 < K; k0 += 32) {
        // A fragment (contiguous, matches packed B): a[0..15]=K kb..kb+15
        bf16x8 alo = *(const bf16x8*)&Atile[arow * K + k0 + koff];
        bf16x8 ahi = *(const bf16x8*)&Atile[arow * K + k0 + koff + 8];
        v16bf av;
#pragma unroll
        for (int i = 0; i < 8; ++i) { av[i] = alo[i]; av[8 + i] = ahi[i]; }
#pragma unroll
        for (int a = 0; a < NACC; ++a) {
            size_t fi = ((size_t)(k0 >> 5) * (NOUT / 16) + (colbase >> 4) + a) * 32 + lane;
            v16bf bv = *(const v16bf*)(Bp + fi * 16);
            acc[a] = __builtin_amdgcn_wmma_f32_16x16x32_bf16(
                false, av, false, bv, (short)0, acc[a], false, false);
        }
    }

    // ---- epilogue ----
#pragma unroll
    for (int a = 0; a < NACC; ++a) {
        float b = bias[colbase + a * 16 + (lane & 15)];
#pragma unroll
        for (int r = 0; r < 8; ++r) acc[a][r] += b;
    }

    if (lng) {
        float srow[8], s2row[8];
#pragma unroll
        for (int r = 0; r < 8; ++r) {
            float s = 0.0f, s2 = 0.0f;
#pragma unroll
            for (int a = 0; a < NACC; ++a) { float v = acc[a][r]; s += v; s2 += v * v; }
#pragma unroll
            for (int m = 1; m < 16; m <<= 1) {
                s  += __shfl_xor(s,  m, 32);
                s2 += __shfl_xor(s2, m, 32);
            }
            srow[r] = s; s2row[r] = s2;
        }
        if ((lane & 15) == 0) {
#pragma unroll
            for (int r = 0; r < 8; ++r) {
                int row = (lane >> 4) * 8 + r;
                red[0][row][wave] = srow[r];
                red[1][row][wave] = s2row[r];
            }
        }
        __syncthreads();
        float mean[8], rstd[8];
#pragma unroll
        for (int r = 0; r < 8; ++r) {
            int row = (lane >> 4) * 8 + r;
            float s = 0.0f, s2 = 0.0f;
#pragma unroll
            for (int w = 0; w < 8; ++w) { s += red[0][row][w]; s2 += red[1][row][w]; }
            float m = s / (float)NOUT;
            float var = s2 / (float)NOUT - m * m;
            mean[r] = m;
            rstd[r] = rsqrtf(var + 1e-5f);
        }
#pragma unroll
        for (int a = 0; a < NACC; ++a) {
            int col = colbase + a * 16 + (lane & 15);
            float g = lng[col], bb = lnb[col];
#pragma unroll
            for (int r = 0; r < 8; ++r)
                acc[a][r] = (acc[a][r] - mean[r]) * rstd[r] * g + bb;
        }
    }

    if (do_gelu) {
#pragma unroll
        for (int a = 0; a < NACC; ++a)
#pragma unroll
            for (int r = 0; r < 8; ++r) acc[a][r] = gelu_exact(acc[a][r]);
    }

#pragma unroll
    for (int a = 0; a < NACC; ++a) {
        int col = colbase + a * 16 + (lane & 15);
#pragma unroll
        for (int r = 0; r < 8; ++r) {
            int row = row0 + (lane >> 4) * 8 + r;
            float v = acc[a][r];
            if (resid)    v += resid[(size_t)row * NOUT + col];
            if (out_f32)  out_f32[(size_t)row * NOUT + col] = v;
            if (out_bf16) out_bf16[(size_t)row * NOUT + col] = (bf16)v;
        }
    }
}

// ---------------------------------------------------------------------------
extern "C" void kernel_launch(void* const* d_in, const int* in_sizes, int n_in,
                              void* d_out, int out_size, void* d_ws, size_t ws_size,
                              hipStream_t stream) {
    const int*   x        = (const int*)d_in[0];
    const int*   ei       = (const int*)d_in[1];
    const int*   ea       = (const int*)d_in[2];
    const int*   batch    = (const int*)d_in[3];
    const float* atom_emb = (const float*)d_in[4];
    const float* vn_emb   = (const float*)d_in[5];
    const float* bond_emb = (const float*)d_in[6];
    const float* eps      = (const float*)d_in[7];
    const float* c_w1 = (const float*)d_in[8];
    const float* c_b1 = (const float*)d_in[9];
    const float* c_lng = (const float*)d_in[10];
    const float* c_lnb = (const float*)d_in[11];
    const float* c_w2 = (const float*)d_in[12];
    const float* c_b2 = (const float*)d_in[13];
    const float* n_g = (const float*)d_in[14];
    const float* n_b = (const float*)d_in[15];
    const float* v_w1 = (const float*)d_in[16];
    const float* v_b1 = (const float*)d_in[17];
    const float* v_lng = (const float*)d_in[18];
    const float* v_lnb = (const float*)d_in[19];
    const float* v_w2 = (const float*)d_in[20];
    const float* v_b2 = (const float*)d_in[21];
    const float* p_w1 = (const float*)d_in[22];
    const float* p_b1 = (const float*)d_in[23];
    const float* p_lng = (const float*)d_in[24];
    const float* p_lnb = (const float*)d_in[25];
    const float* p_w2 = (const float*)d_in[26];
    const float* p_b2 = (const float*)d_in[27];
    float* zout = (float*)d_out;

    // ---- workspace layout ----
    char* ws = (char*)d_ws;
    size_t off = 0;
    auto take = [&](size_t bytes) -> char* {
        char* p = ws + off;
        off = (off + bytes + 255) & ~(size_t)255;
        return p;
    };
    float* h    = (float*)take((size_t)NN * H * 4);
    float* h_in = (float*)take((size_t)NN * H * 4);
    float* aggr = (float*)take((size_t)NN * H * 4);
    bf16*  xb   = (bf16*) take((size_t)NN * H * 2);
    bf16*  t1   = (bf16*) take((size_t)NN * H4 * 2);
    float* vn   = (float*)take((size_t)GG * H * 4);
    bf16*  vpb  = (bf16*) take((size_t)GG * H * 2);
    bf16*  vt1  = (bf16*) take((size_t)GG * H4 * 2);
    bf16*  hgb  = (bf16*) take((size_t)GG * H * 2);
    bf16*  zb   = (bf16*) take((size_t)GG * H * 2);
    float* z    = (float*)take((size_t)GG * H * 4);
    bf16* cw1p = (bf16*)take((size_t)L_LAYERS * H * H4 * 2);
    bf16* cw2p = (bf16*)take((size_t)L_LAYERS * H4 * H * 2);
    bf16* vw1p = (bf16*)take((size_t)(L_LAYERS - 1) * H * H4 * 2);
    bf16* vw2p = (bf16*)take((size_t)(L_LAYERS - 1) * H4 * H * 2);
    bf16* pw1p = (bf16*)take((size_t)H * H * 2);
    bf16* pw2p = (bf16*)take((size_t)H * H * 2);
    (void)ws_size; (void)in_sizes; (void)n_in; (void)out_size;

    // ---- pack weights into WMMA B-fragment layout (bf16) ----
    auto pack = [&](const float* W, bf16* out, int K, int Nout) {
        int total = (K / 32) * (Nout / 16) * 32;
        pack_b_kernel<<<(total + 255) / 256, 256, 0, stream>>>(W, out, K, Nout);
    };
    for (int l = 0; l < L_LAYERS; ++l) {
        pack(c_w1 + (size_t)l * H * H4, cw1p + (size_t)l * H * H4, H, H4);
        pack(c_w2 + (size_t)l * H4 * H, cw2p + (size_t)l * H4 * H, H4, H);
    }
    for (int l = 0; l < L_LAYERS - 1; ++l) {
        pack(v_w1 + (size_t)l * H * H4, vw1p + (size_t)l * H * H4, H, H4);
        pack(v_w2 + (size_t)l * H4 * H, vw2p + (size_t)l * H4 * H, H4, H);
    }
    pack(p_w1, pw1p, H, H);
    pack(p_w2, pw2p, H, H);

    // ---- init ----
    init_h_kernel<<<NN, H, 0, stream>>>(x, atom_emb, h);
    init_vn_kernel<<<GG, H, 0, stream>>>(vn_emb, vn);

    // ---- layers ----
    for (int l = 0; l < L_LAYERS; ++l) {
        addvn_kernel<<<NN, H, 0, stream>>>(h, vn, batch, h_in, aggr);
        edge_kernel<<<EE, H, 0, stream>>>(h_in, ei, ea,
                                          bond_emb + (size_t)l * 5 * H, aggr);
        prepx_kernel<<<NN, H, 0, stream>>>(h_in, aggr, eps, l, xb);

        // GINConv MLP: [N,H] -> +b1 -> LN -> GELU -> bf16 [N,4H]
        fused_gemm<8><<<NN / 16, 256, 0, stream>>>(
            xb, cw1p + (size_t)l * H * H4,
            c_b1 + (size_t)l * H4, c_lng + (size_t)l * H4, c_lnb + (size_t)l * H4,
            1, nullptr, nullptr, t1, H);
        // [N,4H] -> +b2 -> LN(n_g,n_b) -> GELU(l<4) -> +h_in -> h (fp32)
        fused_gemm<2><<<NN / 16, 256, 0, stream>>>(
            t1, cw2p + (size_t)l * H4 * H,
            c_b2 + (size_t)l * H, n_g + (size_t)l * H, n_b + (size_t)l * H,
            (l < L_LAYERS - 1) ? 1 : 0, h_in, h, nullptr, H4);

        if (l < L_LAYERS - 1) {
            segmax_kernel<<<GG, H, 0, stream>>>(h_in, vpb);
            fused_gemm<8><<<GG / 16, 256, 0, stream>>>(
                vpb, vw1p + (size_t)l * H * H4,
                v_b1 + (size_t)l * H4, v_lng + (size_t)l * H4, v_lnb + (size_t)l * H4,
                1, nullptr, nullptr, vt1, H);
            // vn = vn + mlp_out (no outer LN / gelu)
            fused_gemm<2><<<GG / 16, 256, 0, stream>>>(
                vt1, vw2p + (size_t)l * H4 * H,
                v_b2 + (size_t)l * H, nullptr, nullptr,
                0, vn, vn, nullptr, H4);
        }
    }

    // ---- readout + projection head ----
    segsum_kernel<<<GG, H, 0, stream>>>(h, hgb);
    fused_gemm<2><<<GG / 16, 256, 0, stream>>>(
        hgb, pw1p, p_b1, p_lng, p_lnb, 1, nullptr, nullptr, zb, H);
    fused_gemm<2><<<GG / 16, 256, 0, stream>>>(
        zb, pw2p, p_b2, nullptr, nullptr, 0, nullptr, z, nullptr, H);
    l2norm_kernel<<<GG, H, 0, stream>>>(z, zout);
}
